// TransformerBlock_36369783063351
// MI455X (gfx1250) — matrix operand order repaired and checked by reading, not compiled, not run on hardware
//
#include <hip/hip_runtime.h>
#include <hip/hip_bf16.h>
#include <cstdint>

typedef _Float16 half_t;
typedef __attribute__((ext_vector_type(16))) _Float16 v16h;
typedef __attribute__((ext_vector_type(8)))  float    v8f;
typedef __attribute__((ext_vector_type(4)))  float    f4;
typedef __attribute__((ext_vector_type(4)))  int      v4i;

#define D_MODEL 512
#define NTOK    5760
#define NHEAD   4
#define DHEAD   128
#define DH_FFN  1960
#define DH_PAD  2048
#define LOG2E   1.44269504088896f
#define QSCALE  0.088388347648318447f   // 1/sqrt(128)

// ------------- async direct-to-LDS copy (CDNA5 ASYNCcnt path) ---------------
#if defined(__HIP_DEVICE_COMPILE__) && \
    __has_builtin(__builtin_amdgcn_global_load_async_to_lds_b128)
#define HAS_ASYNC 1
#else
#define HAS_ASYNC 0
#endif

#if HAS_ASYNC
typedef __attribute__((address_space(1))) v4i* gp4_t;   // global int4*
typedef __attribute__((address_space(3))) v4i* lp4_t;   // LDS int4*
#endif

__device__ __forceinline__ void async_ld16(void* lds, const void* gl) {
#if HAS_ASYNC
  // flat->as(3): LDS flat addresses carry the byte offset in the low 32 bits
  gp4_t gp = (gp4_t)(uintptr_t)gl;
  lp4_t lp = (lp4_t)(unsigned)(uintptr_t)lds;
  __builtin_amdgcn_global_load_async_to_lds_b128(gp, lp, 0, 0);
#else
  *(f4*)lds = *(const f4*)gl;
#endif
}

// ------------- WMMA fragment helpers (CDNA5 wave32 layouts, ISA 7.12.2) -----
__device__ __forceinline__ v8f wmma_f32(v16h a, v16h b, v8f c) {
  return __builtin_amdgcn_wmma_f32_16x16x32_f16(false, a, false, b, (short)0, c,
                                                false, false);
}

// A 16x32 f16 frag: lane(<16) row m holds K {0..7,16..23}, lane(>=16) {8..15,24..31}
__device__ __forceinline__ v16h frag_a16x32(const half_t* base, int stride) {
  const int lane = threadIdx.x & 31;
  const int hid  = lane >> 4;
  const int m    = lane & 15;
  const half_t* rp = base + m * stride + hid * 8;
  union { v16h v; unsigned u[8]; } f;
#pragma unroll
  for (int vv = 0; vv < 8; ++vv) {
    const int kk = ((vv < 4) ? 0 : 16) + (vv & 3) * 2;
    f.u[vv] = *(const unsigned*)(rp + kk);
  }
  return f.v;
}

// B 32x16 f16 frag: lane holds column n=lane&15, K {0..7,16..23} (+8 for hid=1)
__device__ __forceinline__ v16h frag_b32x16(const half_t* base, int stride) {
  const int lane = threadIdx.x & 31;
  const int hid  = lane >> 4;
  const int n    = lane & 15;
  const half_t* rp = base + n * stride + hid * 8;
  union { v16h v; f4 f[2]; } fb;
  fb.f[0] = *(const f4*)(rp);
  fb.f[1] = *(const f4*)(rp + 16);
  return fb.v;
}

// ------------- generic WMMA GEMM: C[M,N] = A[M,K] * B[N,K]^T ----------------
// 128x128 tile / WG, 8 waves each 32x64 (2x4 WMMA accs), double-buffered LDS
// with async global->LDS copies (last iteration peeled: branch-free hot loop).
#define BM 128
#define BN 128
#define BK 32
#define LDS_K (BK + 8)

__global__ __launch_bounds__(256)
void gemm_kernel(const half_t* __restrict__ A, const half_t* __restrict__ B,
                 const float* __restrict__ bias, int nbias,
                 const float* __restrict__ resid,
                 float* __restrict__ outF, half_t* __restrict__ outH,
                 int N, int K, float scale,
                 size_t bz, size_t biasz, size_t oz) {
  __shared__ __align__(16) half_t As[2][BM][LDS_K];
  __shared__ __align__(16) half_t Bs[2][BN][LDS_K];
  const int z = blockIdx.z;
  B += bz * z;
  bias += biasz * z;
  if (outH) outH += oz * z;
  if (outF) outF += oz * z;

  const int tid = threadIdx.x;
  const int w   = tid >> 5;
  const int wm  = w & 3;   // 4 waves along M (32 rows each)
  const int wn  = w >> 2;  // 2 waves along N (64 cols each)
  const int m0  = blockIdx.y * BM;
  const int n0  = blockIdx.x * BN;
  v8f acc[2][4] = {};

  // tile loader: A 128x32 and B 128x32 halves = 2x512 16B vectors, 4/thread
  auto load_tiles = [&](int buf, int k0) {
#pragma unroll
    for (int i = 0; i < 2; ++i) {
      const int vv = tid * 2 + i;
      const int r = vv >> 2, c8 = vv & 3;
      async_ld16(&As[buf][r][c8 * 8], &A[(size_t)(m0 + r) * K + k0 + c8 * 8]);
      async_ld16(&Bs[buf][r][c8 * 8], &B[(size_t)(n0 + r) * K + k0 + c8 * 8]);
    }
  };
  // 8 WMMAs on the tile currently resident in buffer `buf`
  auto compute = [&](int buf) {
    const half_t* ab = &As[buf][wm * 32][0];
    const v16h a0 = frag_a16x32(ab, LDS_K);
    const v16h a1 = frag_a16x32(ab + 16 * LDS_K, LDS_K);
#pragma unroll
    for (int ni = 0; ni < 4; ++ni) {
      const v16h b = frag_b32x16(&Bs[buf][wn * 64 + ni * 16][0], LDS_K);
      acc[0][ni] = wmma_f32(a0, b, acc[0][ni]);
      acc[1][ni] = wmma_f32(a1, b, acc[1][ni]);
    }
  };

  load_tiles(0, 0);
  int buf = 0;
  for (int k0 = 0; k0 + BK < K; k0 += BK) {      // steady state: branch-free
    __syncthreads();                             // prev readers of buf^1 done
    load_tiles(buf ^ 1, k0 + BK);                // prefetch next tile
#if HAS_ASYNC
    asm volatile("s_wait_asynccnt 4" ::: "memory");  // tile `buf` landed
#endif
    __syncthreads();                             // all waves' tile data visible
    compute(buf);
    buf ^= 1;
  }
  __syncthreads();                               // epilogue: last tile
#if HAS_ASYNC
  asm volatile("s_wait_asynccnt 0" ::: "memory");
#endif
  __syncthreads();
  compute(buf);

  const int lane = tid & 31;
  const int hid = lane >> 4, nn = lane & 15;
#pragma unroll
  for (int mi = 0; mi < 2; ++mi)
#pragma unroll
    for (int ni = 0; ni < 4; ++ni) {
      const int gcol = n0 + wn * 64 + ni * 16 + nn;
      const float bv = (gcol < nbias) ? bias[gcol] : 0.f;
#pragma unroll
      for (int r = 0; r < 8; ++r) {
        const int grow = m0 + wm * 32 + mi * 16 + r + hid * 8;
        float val = (acc[mi][ni][r] + bv) * scale;
        if (resid) val += resid[(size_t)grow * N + gcol];
        if (outF)  outF[(size_t)grow * N + gcol] = val;
        else       outH[(size_t)grow * N + gcol] = (half_t)val;
      }
    }
}

// ------------- flash attention (online softmax, 32-key tiles) ---------------
// grid = (NTOK/128, NHEAD), 256 threads; score scale applied post-WMMA.
__global__ __launch_bounds__(256)
void attn_kernel(const half_t* __restrict__ q, const half_t* __restrict__ k,
                 const half_t* __restrict__ v, half_t* __restrict__ o) {
  __shared__ __align__(16) half_t Ks[32][136];    // [key][d]
  __shared__ __align__(16) half_t Vst[128][40];   // [d][key] (transposed)
  __shared__ __align__(16) half_t Ps[8][16][40];  // per-wave P staging
  const int tid = threadIdx.x;
  const int w    = tid >> 5;
  const int lane = tid & 31;
  const int hid  = lane >> 4, nn = lane & 15;
  const int head = blockIdx.y;
  const int q0   = blockIdx.x * 128 + w * 16;

  v16h aq[4];                                      // Q resident: 16 x 128
#pragma unroll
  for (int c = 0; c < 4; ++c) {
    union { v16h v; unsigned u[8]; } f;
    const half_t* rp =
        q + (size_t)(q0 + (lane & 15)) * D_MODEL + head * DHEAD + c * 32 + hid * 8;
#pragma unroll
    for (int vv = 0; vv < 8; ++vv) {
      const int kk = ((vv < 4) ? 0 : 16) + (vv & 3) * 2;
      f.u[vv] = *(const unsigned*)(rp + kk);
    }
    aq[c] = f.v;
  }

  v8f oacc[8] = {};
  float mrun[8], lrun[8];
#pragma unroll
  for (int r = 0; r < 8; ++r) { mrun[r] = -1e30f; lrun[r] = 0.f; }

  for (int t0 = 0; t0 < NTOK; t0 += 32) {
    __syncthreads();
#pragma unroll
    for (int i = 0; i < 2; ++i) {                  // K + V(transposed) tiles
      const int vv = tid * 2 + i;
      const int r = vv >> 4, c8 = vv & 15;
      *(f4*)&Ks[r][c8 * 8] =
          *(const f4*)&k[(size_t)(t0 + r) * D_MODEL + head * DHEAD + c8 * 8];
      union { f4 f; half_t h[8]; } tv;
      tv.f = *(const f4*)&v[(size_t)(t0 + r) * D_MODEL + head * DHEAD + c8 * 8];
#pragma unroll
      for (int j = 0; j < 8; ++j) Vst[c8 * 8 + j][r] = tv.h[j];
    }
    __syncthreads();

    v8f s[2] = {};                                 // S = Q K^T (16x32)
#pragma unroll
    for (int j = 0; j < 2; ++j)
#pragma unroll
      for (int c = 0; c < 4; ++c)
        s[j] = wmma_f32(aq[c], frag_b32x16(&Ks[j * 16][c * 32], 136), s[j]);
#pragma unroll
    for (int j = 0; j < 2; ++j)
#pragma unroll
      for (int r = 0; r < 8; ++r) s[j][r] *= QSCALE;

    float alpha[8];
#pragma unroll
    for (int r = 0; r < 8; ++r) {
      float mn = fmaxf(s[0][r], s[1][r]);
#pragma unroll
      for (int msk = 1; msk < 16; msk <<= 1)
        mn = fmaxf(mn, __shfl_xor(mn, msk, 32));
      mn = fmaxf(mn, mrun[r]);
      alpha[r] = exp2f((mrun[r] - mn) * LOG2E);
      const float p0 = exp2f((s[0][r] - mn) * LOG2E);
      const float p1 = exp2f((s[1][r] - mn) * LOG2E);
      float rs = p0 + p1;
#pragma unroll
      for (int msk = 1; msk < 16; msk <<= 1)
        rs += __shfl_xor(rs, msk, 32);
      lrun[r] = lrun[r] * alpha[r] + rs;
      mrun[r] = mn;
      Ps[w][r + hid * 8][nn]      = (half_t)p0;    // restage P in A layout
      Ps[w][r + hid * 8][16 + nn] = (half_t)p1;
    }
#pragma unroll
    for (int cc = 0; cc < 8; ++cc)
#pragma unroll
      for (int r = 0; r < 8; ++r) oacc[cc][r] *= alpha[r];
    __syncthreads();

    const v16h pf = frag_a16x32(&Ps[w][0][0], 40);
#pragma unroll
    for (int cc = 0; cc < 8; ++cc)                 // O += P * V
      oacc[cc] = wmma_f32(pf, frag_b32x16(&Vst[cc * 16][0], 40), oacc[cc]);
  }

#pragma unroll
  for (int r = 0; r < 8; ++r) {
    const float linv = 1.f / lrun[r];
    const int grow = q0 + r + hid * 8;
#pragma unroll
    for (int cc = 0; cc < 8; ++cc)
      o[(size_t)grow * D_MODEL + head * DHEAD + cc * 16 + nn] =
          (half_t)(oacc[cc][r] * linv);
  }
}

// ------------- LayerNorm: fp32 in -> f16 out (one row / block) --------------
__global__ __launch_bounds__(256)
void ln_kernel(const float* __restrict__ x, const float* __restrict__ g,
               const float* __restrict__ b, half_t* __restrict__ out) {
  __shared__ float red[8];
  const int row = blockIdx.x, tid = threadIdx.x;
  const float* xr = x + (size_t)row * D_MODEL;
  const float v0 = xr[tid], v1 = xr[tid + 256];
  float s = v0 + v1;
#pragma unroll
  for (int m = 16; m; m >>= 1) s += __shfl_xor(s, m, 32);
  if ((tid & 31) == 0) red[tid >> 5] = s;
  __syncthreads();
  float t = 0.f;
#pragma unroll
  for (int i = 0; i < 8; ++i) t += red[i];
  const float mean = t * (1.f / D_MODEL);
  __syncthreads();
  const float d0 = v0 - mean, d1 = v1 - mean;
  s = d0 * d0 + d1 * d1;
#pragma unroll
  for (int m = 16; m; m >>= 1) s += __shfl_xor(s, m, 32);
  if ((tid & 31) == 0) red[tid >> 5] = s;
  __syncthreads();
  t = 0.f;
#pragma unroll
  for (int i = 0; i < 8; ++i) t += red[i];
  const float rs = rsqrtf(t * (1.f / D_MODEL) + 1e-5f);
  out[(size_t)row * D_MODEL + tid]       = (half_t)(d0 * rs * g[tid] + b[tid]);
  out[(size_t)row * D_MODEL + tid + 256] = (half_t)(d1 * rs * g[tid + 256] + b[tid + 256]);
}

// ------------- fp32 -> f16 weight convert with zero padding -----------------
__global__ void cvt_pad_kernel(const float* __restrict__ src, int sr, int sc,
                               half_t* __restrict__ dst, int dr, int dc) {
  const size_t i = (size_t)blockIdx.x * 256 + threadIdx.x;
  if (i >= (size_t)dr * dc) return;
  const int r = (int)(i / dc), c = (int)(i % dc);
  const float v = (r < sr && c < sc) ? src[(size_t)r * sc + c] : 0.f;
  dst[i] = (half_t)v;
}

__global__ void pack_bias_kernel(const float* __restrict__ a,
                                 const float* __restrict__ b,
                                 const float* __restrict__ c,
                                 float* __restrict__ dst) {
  const int i = blockIdx.x * 256 + threadIdx.x;
  if (i < D_MODEL) {
    dst[i] = a[i];
    dst[D_MODEL + i] = b[i];
    dst[2 * D_MODEL + i] = c[i];
  }
}

// ------------- fold (adjoint of unfold) + normalize -------------------------
__global__ void fold_kernel(const half_t* __restrict__ y1h, float* __restrict__ nd) {
  const int i = blockIdx.x * 256 + threadIdx.x;   // 8*40*60*108 = 2073600
  if (i >= 8 * 40 * 60 * 108) return;
  const int x = i % 108;
  int t = i / 108;
  const int y = t % 60; t /= 60;
  const int c = t % 40;
  const int b = t / 40;
  float sum = 0.f;
  int den = 0;
  for (int kh = y % 3; kh < 7; kh += 3) {
    const int py = (y + 3 - kh) / 3;
    if (py < 0 || py >= 20) continue;
    for (int kw = x % 3; kw < 7; kw += 3) {
      const int px = (x + 3 - kw) / 3;
      if (px < 0 || px >= 36) continue;
      sum += (float)y1h[(size_t)(b * 720 + py * 36 + px) * DH_PAD +
                        c * 49 + kh * 7 + kw];
      ++den;
    }
  }
  nd[i] = sum / (float)den;
}

// ------------- unfold + ReLU -> f16 (zero K-padding for final GEMM) ---------
__global__ void unfold_relu_kernel(const float* __restrict__ nd,
                                   half_t* __restrict__ y2) {
  const size_t i = (size_t)blockIdx.x * 256 + threadIdx.x;
  if (i >= (size_t)NTOK * DH_PAD) return;
  const int dh = (int)(i % DH_PAD);
  const int row = (int)(i / DH_PAD);
  float val = 0.f;
  if (dh < DH_FFN) {
    const int c = dh / 49, rem = dh % 49, kh = rem / 7, kw = rem % 7;
    const int b = row / 720, p = row % 720, py = p / 36, px = p % 36;
    const int y = py * 3 - 3 + kh, x = px * 3 - 3 + kw;
    if (y >= 0 && y < 60 && x >= 0 && x < 108)
      val = nd[((b * 40 + c) * 60 + y) * 108 + x];
  }
  y2[i] = (half_t)fmaxf(val, 0.f);
}

// ------------- launcher ------------------------------------------------------
extern "C" void kernel_launch(void* const* d_in, const int* in_sizes, int n_in,
                              void* d_out, int out_size, void* d_ws, size_t ws_size,
                              hipStream_t stream) {
  (void)in_sizes; (void)n_in; (void)out_size; (void)ws_size;
  const float* x   = (const float*)d_in[0];
  const float* Wq  = (const float*)d_in[1];
  const float* bq  = (const float*)d_in[2];
  const float* Wk  = (const float*)d_in[3];
  const float* bk  = (const float*)d_in[4];
  const float* Wv  = (const float*)d_in[5];
  const float* bv  = (const float*)d_in[6];
  const float* Wo  = (const float*)d_in[7];
  const float* bo  = (const float*)d_in[8];
  const float* g1  = (const float*)d_in[9];
  const float* be1 = (const float*)d_in[10];
  const float* g2  = (const float*)d_in[11];
  const float* be2 = (const float*)d_in[12];
  const float* W1  = (const float*)d_in[13];
  const float* b1  = (const float*)d_in[14];
  const float* W2  = (const float*)d_in[15];
  const float* b2  = (const float*)d_in[16];
  float* out = (float*)d_out;

  char* p = (char*)d_ws;
  auto alloc = [&](size_t bytes) {
    char* r = p;
    p += (bytes + 255) & ~(size_t)255;
    return (void*)r;
  };
  // NOTE: Wqh/Wkh/Wvh and qh/kh/vh must stay contiguous (gridDim.z=3 GEMM).
  half_t* Wqh = (half_t*)alloc((size_t)512 * 512 * 2);
  half_t* Wkh = (half_t*)alloc((size_t)512 * 512 * 2);
  half_t* Wvh = (half_t*)alloc((size_t)512 * 512 * 2);
  half_t* Woh = (half_t*)alloc((size_t)512 * 512 * 2);
  half_t* W1h = (half_t*)alloc((size_t)DH_PAD * 512 * 2);
  half_t* W2h = (half_t*)alloc((size_t)512 * DH_PAD * 2);
  float*  b3  = (float*) alloc((size_t)3 * 512 * 4);      // packed bq|bk|bv
  half_t* hN  = (half_t*)alloc((size_t)NTOK * 512 * 2);   // LN1 out
  half_t* qh  = (half_t*)alloc((size_t)NTOK * 512 * 2);
  half_t* kh  = (half_t*)alloc((size_t)NTOK * 512 * 2);
  half_t* vh  = (half_t*)alloc((size_t)NTOK * 512 * 2);
  half_t* oh  = (half_t*)alloc((size_t)NTOK * 512 * 2);   // attn out
  float*  x1  = (float*) alloc((size_t)NTOK * 512 * 4);   // residual 1
  half_t* yh  = (half_t*)alloc((size_t)NTOK * 512 * 2);   // LN2 out
  half_t* y1h = (half_t*)alloc((size_t)NTOK * DH_PAD * 2);
  float*  nd  = (float*) alloc((size_t)8 * 40 * 60 * 108 * 4);
  half_t* y2h = (half_t*)alloc((size_t)NTOK * DH_PAD * 2);

  // weight conversion + bias packing
  cvt_pad_kernel<<<(512 * 512 + 255) / 256, 256, 0, stream>>>(Wq, 512, 512, Wqh, 512, 512);
  cvt_pad_kernel<<<(512 * 512 + 255) / 256, 256, 0, stream>>>(Wk, 512, 512, Wkh, 512, 512);
  cvt_pad_kernel<<<(512 * 512 + 255) / 256, 256, 0, stream>>>(Wv, 512, 512, Wvh, 512, 512);
  cvt_pad_kernel<<<(512 * 512 + 255) / 256, 256, 0, stream>>>(Wo, 512, 512, Woh, 512, 512);
  cvt_pad_kernel<<<(DH_PAD * 512 + 255) / 256, 256, 0, stream>>>(W1, DH_FFN, 512, W1h, DH_PAD, 512);
  cvt_pad_kernel<<<(512 * DH_PAD + 255) / 256, 256, 0, stream>>>(W2, 512, DH_FFN, W2h, 512, DH_PAD);
  pack_bias_kernel<<<2, 256, 0, stream>>>(bq, bk, bv, b3);

  // LN1
  ln_kernel<<<NTOK, 256, 0, stream>>>(x, g1, be1, hN);

  // fused QKV projections (z in {q,k,v})
  gemm_kernel<<<dim3(512 / BN, NTOK / BM, 3), 256, 0, stream>>>(
      hN, Wqh, b3, 512, nullptr, nullptr, qh, 512, 512, 1.f,
      (size_t)512 * 512, 512, (size_t)NTOK * 512);

  // flash attention (applies 1/sqrt(dh) to scores)
  attn_kernel<<<dim3(NTOK / 128, NHEAD), 256, 0, stream>>>(qh, kh, vh, oh);

  // O projection + residual -> x1 (fp32)
  gemm_kernel<<<dim3(512 / BN, NTOK / BM), 256, 0, stream>>>(
      oh, Woh, bo, 512, x, x1, nullptr, 512, 512, 1.f, 0, 0, 0);

  // LN2
  ln_kernel<<<NTOK, 256, 0, stream>>>(x1, g2, be2, yh);

  // FFN up: y1 = yh @ W1^T + b1  (N padded to 2048)
  gemm_kernel<<<dim3(DH_PAD / BN, NTOK / BM), 256, 0, stream>>>(
      yh, W1h, b1, DH_FFN, nullptr, nullptr, y1h, DH_PAD, 512, 1.f, 0, 0, 0);

  // fold + normalize, then unfold + ReLU
  fold_kernel<<<(8 * 40 * 60 * 108) / 256, 256, 0, stream>>>(y1h, nd);
  unfold_relu_kernel<<<(int)(((size_t)NTOK * DH_PAD + 255) / 256), 256, 0, stream>>>(nd, y2h);

  // FFN down + residual -> out (fp32)
  gemm_kernel<<<dim3(512 / BN, NTOK / BM), 256, 0, stream>>>(
      y2h, W2h, b2, 512, x1, out, nullptr, 512, DH_PAD, 1.f, 0, 0, 0);
}
